// SelfAttention_46351287058663
// MI455X (gfx1250) — matrix operand order repaired and checked by reading, not compiled, not run on hardware
//
#include <hip/hip_runtime.h>
#include <hip/hip_bf16.h>
#include <cstddef>
#include <cstdint>

// ---------------------------------------------------------------------------
// Self-attention forward for MI455X (gfx1250), wave32 + WMMA f16->f32,
// with TDM (tensor_load_to_lds) double-buffered K/V staging in attention.
// ---------------------------------------------------------------------------

typedef __attribute__((ext_vector_type(16))) _Float16 v16h;
typedef __attribute__((ext_vector_type(8)))  _Float16 v8h;
typedef __attribute__((ext_vector_type(4)))  _Float16 v4h;
typedef __attribute__((ext_vector_type(8)))  float    v8f;
typedef __attribute__((ext_vector_type(4)))  uint32_t u32x4;
typedef __attribute__((ext_vector_type(8)))  int      i32x8;
typedef __attribute__((ext_vector_type(4)))  int      i32x4;

static constexpr int B_   = 2;
static constexpr int T_   = 2048;
static constexpr int C_   = 1024;
static constexpr int H_   = 16;
static constexpr int D_   = 64;
static constexpr int M_   = B_ * T_;   // 4096 GEMM rows

// LDS row strides (halves). 68 -> 34 words/row: 16-row column strides map the
// two lane-halves to disjoint bank sets (544 % 64 == 32). TDM hardware padding
// reproduces this stride: row = 32 DWORDs (pad_interval code 4), pad = 2 DWORDs
// (pad_amount code 1) -> 64 + 4 halves. P buffer stride 36 breaks 8-row
// aliasing (144 % 64 == 16). K/P fragment reads are 64-bit; P stores, V reads 16-bit.
static constexpr int KVS_ = 68;
static constexpr int PS_  = 36;

// ---------------------------------------------------------------------------
// fragment loaders (contiguous k order, k = 16*(lane/16) + s, same for A and B, wave32)
// ---------------------------------------------------------------------------

// A-matrix 16x32 f16: lanes 0-15 = rows 0-15 holding K 0..15;
// lanes 16-31 = rows 0-15 holding K 16..31. 2 halves packed/VGPR.
__device__ __forceinline__ v16h load_a16(const _Float16* base, int ld,
                                         int m0, int k0, int lane) {
  const int lr = lane & 15, hf = lane >> 4;
  const _Float16* p = base + (size_t)(m0 + lr) * ld + k0 + hf * 16;
  v8h lo = *(const v8h*)(p);
  v8h hi = *(const v8h*)(p + 8);
  v16h a;
#pragma unroll
  for (int i = 0; i < 8; ++i) { a[i] = lo[i]; a[i + 8] = hi[i]; }
  return a;
}

// B-matrix 32x16 f16 (B[k][n] = W[n][k], W row-major NxK):
// lanes 0-15 = col n holding K 0..15; lanes 16-31 = col n holding K 16..31.
__device__ __forceinline__ v16h load_b16(const _Float16* base, int ld,
                                         int n0, int k0, int lane) {
  const int lr = lane & 15, hf = lane >> 4;
  const _Float16* p = base + (size_t)(n0 + lr) * ld + k0 + hf * 16;
  v8h lo = *(const v8h*)(p);
  v8h hi = *(const v8h*)(p + 8);
  v16h b;
#pragma unroll
  for (int i = 0; i < 8; ++i) { b[i] = lo[i]; b[i + 8] = hi[i]; }
  return b;
}

__device__ __forceinline__ v8f wmma32(v16h a, v16h b, v8f c) {
  return __builtin_amdgcn_wmma_f32_16x16x32_f16(false, a, false, b,
                                                (short)0, c, false, false);
}

// ---------------------------------------------------------------------------
// TDM: issue a 32-row x 64-half tile DMA (global -> LDS) with hardware row
// padding to stride 68 halves. Descriptor per CDNA5 ISA.
// This toolchain (clang-23 / therock-10.0) uses the 6-arg builtin form.
// ---------------------------------------------------------------------------
__device__ __forceinline__ void tdm_load_tile(const _Float16* gsrc,
                                              uint32_t lds_addr) {
  const uint64_t ga = (uint64_t)(uintptr_t)gsrc;
  // group0: count=1 | lds_addr | global_addr[56:0] | type=2
  u32x4 g0 = { 1u,
               lds_addr,
               (uint32_t)ga,
               (2u << 30) | (uint32_t)((ga >> 32) & 0x1FFFFFFu) };
  // group1: data_size=1 (2B), pad_enable, pad_interval=4 (32 DW), pad_amount=1
  //         (2 DW); tensor_dim0=64, tensor_dim1=32; tile 64 x 32;
  //         tensor_dim0_stride = 1024 (elements).
  i32x8 g1 = { (int)((1u << 16) | (1u << 20) | (4u << 22) | (1u << 25)),
               (int)(64u << 16),          // tensor_dim0[15:0] << 16
               (int)(32u << 16),          // tensor_dim0[31:16]=0 | td1 low
               (int)(64u << 16),          // td1 high = 0 | tile_dim0 = 64
               (int)32,                   // tile_dim1 = 32 | tile_dim2 = 0
               (int)1024,                 // tensor_dim0_stride[31:0]
               0, 0 };
  i32x4 z4 = { 0, 0, 0, 0 };              // groups 2/3 unused (2-D tile)
  i32x8 z8 = { 0, 0, 0, 0, 0, 0, 0, 0 };
  __builtin_amdgcn_tensor_load_to_lds(g0, g1, z4, z4, z8, 0);
}

// ---------------------------------------------------------------------------
// f32 -> f16 conversion (scale folds the attention 1/sqrt(D) into Wq)
// ---------------------------------------------------------------------------
__global__ __launch_bounds__(256)
void k_cvt(const float* __restrict__ src, _Float16* __restrict__ dst,
           int n, float scale) {
  int i = blockIdx.x * 256 + threadIdx.x;
  if (i < n) dst[i] = (_Float16)(src[i] * scale);
}

// ---------------------------------------------------------------------------
// GEMM: Out[M x N] = A[M x K] * W[N x K]^T   (all f16 in, f16 or f32 out)
// block = 256 threads = 8 waves; block tile 128x128; wave tile 32x64.
// ---------------------------------------------------------------------------
__device__ __forceinline__ void gemm_core(const _Float16* __restrict__ A,
                                          const _Float16* __restrict__ W,
                                          int m0, int n0, int lane,
                                          v8f acc[2][4]) {
  constexpr int K = C_;
  for (int kk = 0; kk < K; kk += 32) {
    v16h af[2], bf[4];
#pragma unroll
    for (int i = 0; i < 2; ++i) af[i] = load_a16(A, K, m0 + 16 * i, kk, lane);
#pragma unroll
    for (int j = 0; j < 4; ++j) bf[j] = load_b16(W, K, n0 + 16 * j, kk, lane);
#pragma unroll
    for (int i = 0; i < 2; ++i)
#pragma unroll
      for (int j = 0; j < 4; ++j) acc[i][j] = wmma32(af[i], bf[j], acc[i][j]);
  }
}

__global__ __launch_bounds__(256)
void k_gemm_h(const _Float16* __restrict__ A, const _Float16* __restrict__ W,
              _Float16* __restrict__ Out) {
  const int lane = threadIdx.x & 31, wave = threadIdx.x >> 5;
  const int wm = wave & 3, wn = wave >> 2;
  const int m0 = blockIdx.x * 128 + wm * 32;
  const int n0 = blockIdx.y * 128 + wn * 64;
  v8f acc[2][4] = {};
  gemm_core(A, W, m0, n0, lane, acc);
  const int lr = lane & 15, hf = lane >> 4;
#pragma unroll
  for (int i = 0; i < 2; ++i)
#pragma unroll
    for (int j = 0; j < 4; ++j)
#pragma unroll
      for (int r = 0; r < 8; ++r) {
        int row = m0 + 16 * i + r + 8 * hf;
        int col = n0 + 16 * j + lr;
        Out[(size_t)row * C_ + col] = (_Float16)acc[i][j][r];
      }
}

__global__ __launch_bounds__(256)
void k_gemm_f_bias(const _Float16* __restrict__ A, const _Float16* __restrict__ W,
                   const float* __restrict__ bias, float* __restrict__ Out) {
  const int lane = threadIdx.x & 31, wave = threadIdx.x >> 5;
  const int wm = wave & 3, wn = wave >> 2;
  const int m0 = blockIdx.x * 128 + wm * 32;
  const int n0 = blockIdx.y * 128 + wn * 64;
  v8f acc[2][4] = {};
  gemm_core(A, W, m0, n0, lane, acc);
  const int lr = lane & 15, hf = lane >> 4;
#pragma unroll
  for (int i = 0; i < 2; ++i)
#pragma unroll
    for (int j = 0; j < 4; ++j) {
      int col = n0 + 16 * j + lr;
      float bv = bias[col];
#pragma unroll
      for (int r = 0; r < 8; ++r) {
        int row = m0 + 16 * i + r + 8 * hf;
        Out[(size_t)row * C_ + col] = acc[i][j][r] + bv;
      }
    }
}

// ---------------------------------------------------------------------------
// Flash attention (causal). Grid (T/128, H, B); block 256 threads = 8 waves.
// Wave handles 16 query rows. K/V tiles (32 keys x 64) are DMA'd into LDS by
// the TDM (double-buffered, TENSORcnt-tracked); the next tile's DMA overlaps
// the current tile's WMMA + softmax work.
// ---------------------------------------------------------------------------
__global__ __launch_bounds__(256)
void k_attn(const _Float16* __restrict__ Q, const _Float16* __restrict__ K,
            const _Float16* __restrict__ V, _Float16* __restrict__ O) {
  const int tid  = threadIdx.x;
  const int lane = tid & 31, wave = tid >> 5;
  const int lr = lane & 15, hf = lane >> 4;
  const int h = blockIdx.y, b = blockIdx.z;
  const int q0 = blockIdx.x * 128 + wave * 16;  // wave's query-row base

  const size_t headoff = (size_t)b * T_ * C_ + h * D_;
  const _Float16* Qh = Q + headoff;
  const _Float16* Kh = K + headoff;
  const _Float16* Vh = V + headoff;

  __shared__ _Float16 Kt[2][32 * KVS_];
  __shared__ _Float16 Vt[2][32 * KVS_];
  __shared__ _Float16 pbuf[8][16 * PS_];

  // Q fragments, resident for whole kernel (K-dim = D = 64 -> 2 frags)
  v16h aq0 = load_a16(Qh, C_, q0, 0, lane);
  v16h aq1 = load_a16(Qh, C_, q0, 32, lane);

  v8f o[4] = {};                      // 16 x 64 output accumulator
  float mrow[8], lrow[8];
#pragma unroll
  for (int r = 0; r < 8; ++r) { mrow[r] = -1e30f; lrow[r] = 0.f; }

  const int jjmax = blockIdx.x * 4 + 3;          // inclusive

  // prologue: DMA tile 0 into buffer 0
  if (wave == 0) {
    tdm_load_tile(Kh, (uint32_t)(uintptr_t)&Kt[0][0]);
    tdm_load_tile(Vh, (uint32_t)(uintptr_t)&Vt[0][0]);
  }

  for (int jj = 0; jj <= jjmax; ++jj) {
    const int kb  = jj * 32;
    const int cur = jj & 1;
    const bool active = (kb <= q0 + 15);

    if (wave == 0) __builtin_amdgcn_s_wait_tensorcnt(0);
    __syncthreads();                  // current buffer visible to all waves
    if (wave == 0 && jj < jjmax) {    // overlap next tile DMA with compute
      const size_t goff = (size_t)(kb + 32) * C_;
      tdm_load_tile(Kh + goff, (uint32_t)(uintptr_t)&Kt[cur ^ 1][0]);
      tdm_load_tile(Vh + goff, (uint32_t)(uintptr_t)&Vt[cur ^ 1][0]);
    }

    if (active) {
      // ---- S = Q K^T (16 x 32) from LDS K tile ----
      v8f s[2];
#pragma unroll
      for (int nf = 0; nf < 2; ++nf) {
        v16h bk0, bk1;
        const _Float16* kr = &Kt[cur][(nf * 16 + lr) * KVS_ + hf * 16];
#pragma unroll
        for (int j = 0; j < 4; ++j) {
          v4h c0 = *(const v4h*)(kr + 4 * j);        // k-step 0 (d 0..31)
          v4h c1 = *(const v4h*)(kr + 32 + 4 * j);   // k-step 1 (d 32..63)
#pragma unroll
          for (int e = 0; e < 4; ++e) { bk0[4 * j + e] = c0[e]; bk1[4 * j + e] = c1[e]; }
        }
        v8f z = {};
        z = wmma32(aq0, bk0, z);
        s[nf] = wmma32(aq1, bk1, z);
      }
      // ---- causal mask ----
#pragma unroll
      for (int nf = 0; nf < 2; ++nf) {
        int col = kb + 16 * nf + lr;
#pragma unroll
        for (int r = 0; r < 8; ++r) {
          int row = q0 + r + 8 * hf;
          if (col > row) s[nf][r] = -1e30f;
        }
      }
      // ---- online softmax (row stats across 16-lane halves) ----
      float alpha[8];
#pragma unroll
      for (int r = 0; r < 8; ++r) {
        float rm = fmaxf(s[0][r], s[1][r]);
#pragma unroll
        for (int off = 1; off < 16; off <<= 1)
          rm = fmaxf(rm, __shfl_xor(rm, off, 32));
        float mn = fmaxf(mrow[r], rm);
        alpha[r] = __expf(mrow[r] - mn);
        s[0][r] = __expf(s[0][r] - mn);
        s[1][r] = __expf(s[1][r] - mn);
        float rs = s[0][r] + s[1][r];
#pragma unroll
        for (int off = 1; off < 16; off <<= 1)
          rs += __shfl_xor(rs, off, 32);
        lrow[r] = lrow[r] * alpha[r] + rs;
        mrow[r] = mn;
      }
      // rescale running output
#pragma unroll
      for (int dt = 0; dt < 4; ++dt)
#pragma unroll
        for (int r = 0; r < 8; ++r) o[dt][r] *= alpha[r];
      // stage P (C-layout -> row-major LDS tile, per-wave region)
#pragma unroll
      for (int nf = 0; nf < 2; ++nf)
#pragma unroll
        for (int r = 0; r < 8; ++r)
          pbuf[wave][(r + 8 * hf) * PS_ + 16 * nf + lr] = (_Float16)s[nf][r];
      // reload P in A-fragment layout (same wave: LDS ops are in-order)
      v16h pa;
      const _Float16* pr = &pbuf[wave][lr * PS_ + hf * 16];
#pragma unroll
      for (int j = 0; j < 2; ++j) {
        v4h c0 = *(const v4h*)(pr + 4 * j);
        v4h c1 = *(const v4h*)(pr + 8 + 4 * j);
#pragma unroll
        for (int e = 0; e < 4; ++e) { pa[4 * j + e] = c0[e]; pa[8 + 4 * j + e] = c1[e]; }
      }
      // ---- O += P V : B[k=key][n=d], column reads from LDS V tile ----
#pragma unroll
      for (int dt = 0; dt < 4; ++dt) {
        const _Float16* vb = &Vt[cur][(hf * 16) * KVS_ + dt * 16 + lr];
        v16h bv;
#pragma unroll
        for (int i = 0; i < 16; ++i) bv[i] = vb[i * KVS_];
        o[dt] = wmma32(pa, bv, o[dt]);
      }
    }
  }

  // ---- epilogue: O / l, store f16 into [B*T][C] at head window ----
  float inv[8];
#pragma unroll
  for (int r = 0; r < 8; ++r) inv[r] = 1.0f / lrow[r];
  _Float16* Oh = O + headoff;
#pragma unroll
  for (int dt = 0; dt < 4; ++dt)
#pragma unroll
    for (int r = 0; r < 8; ++r) {
      int row = q0 + r + 8 * hf;
      Oh[(size_t)row * C_ + dt * 16 + lr] = (_Float16)(o[dt][r] * inv[r]);
    }
}

// ---------------------------------------------------------------------------
// host launcher
// ---------------------------------------------------------------------------
extern "C" void kernel_launch(void* const* d_in, const int* in_sizes, int n_in,
                              void* d_out, int out_size, void* d_ws, size_t ws_size,
                              hipStream_t stream) {
  const float* x  = (const float*)d_in[0];
  const float* Wq = (const float*)d_in[1];
  const float* Wk = (const float*)d_in[2];
  const float* Wv = (const float*)d_in[3];
  const float* Wo = (const float*)d_in[4];
  const float* bo = (const float*)d_in[5];
  float* out = (float*)d_out;

  _Float16* ws = (_Float16*)d_ws;
  const size_t NX = (size_t)M_ * C_;      // 4 M elements
  const size_t NW = (size_t)C_ * C_;      // 1 M elements
  _Float16* xh  = ws;
  _Float16* wqh = xh  + NX;
  _Float16* wkh = wqh + NW;
  _Float16* wvh = wkh + NW;
  _Float16* woh = wvh + NW;
  _Float16* qh  = woh + NW;
  _Float16* kh  = qh  + NX;
  _Float16* vh  = kh  + NX;
  _Float16* ah  = vh  + NX;

  const float qscale = 0.125f;            // D^-0.5 = 64^-0.5 folded into Wq

  k_cvt<<<dim3((NX + 255) / 256), 256, 0, stream>>>(x,  xh,  (int)NX, 1.0f);
  k_cvt<<<dim3((NW + 255) / 256), 256, 0, stream>>>(Wq, wqh, (int)NW, qscale);
  k_cvt<<<dim3((NW + 255) / 256), 256, 0, stream>>>(Wk, wkh, (int)NW, 1.0f);
  k_cvt<<<dim3((NW + 255) / 256), 256, 0, stream>>>(Wv, wvh, (int)NW, 1.0f);
  k_cvt<<<dim3((NW + 255) / 256), 256, 0, stream>>>(Wo, woh, (int)NW, 1.0f);

  dim3 ggrid(M_ / 128, C_ / 128);         // 32 x 8
  k_gemm_h<<<ggrid, 256, 0, stream>>>(xh, wqh, qh);
  k_gemm_h<<<ggrid, 256, 0, stream>>>(xh, wkh, kh);
  k_gemm_h<<<ggrid, 256, 0, stream>>>(xh, wvh, vh);

  k_attn<<<dim3(T_ / 128, H_, B_), 256, 0, stream>>>(qh, kh, vh, ah);

  k_gemm_f_bias<<<ggrid, 256, 0, stream>>>(ah, woh, bo, out);
}
